// LambdaLayer_27650999452136
// MI455X (gfx1250) — compile-verified, matrix operand order checked by reading, not run on hardware
//
// LambdaLayer for MI455X (gfx1250, wave32, WMMA bf16, async LDS copies).
//
// Refactor: position term computed attention-style:
//   S[b,h,n,m] = sum_k q[b,h,n,k]*E[n,m,k];  out += S @ values   (19.3 GFLOP)
// instead of the reference's bnkv einsum (68.7 GFLOP). E (268 MB f32) is the
// memory floor: ~11.5 us @ 23.3 TB/s. bf16 WMMA (16x16x32) keeps compute far
// under that; f32 WMMA (16x16x4) would not. E is converted f32->bf16 in-register.

#include <hip/hip_runtime.h>
#include <stdint.h>

#define USE_ASYNC_LDS 1

typedef __attribute__((ext_vector_type(16))) __bf16        v16bf;
typedef __attribute__((ext_vector_type(8)))  float         v8f;
typedef __attribute__((ext_vector_type(4)))  float         f32x4;
typedef __attribute__((ext_vector_type(4)))  unsigned int  u32x4;

#define B_  4
#define N_  2048
#define M_  2048
#define D_  512
#define K_  16
#define V_  128
#define H_  4
#define MSPLIT 2

union BF16V {
  v16bf  v;
  __bf16 e[16];
  u32x4  q[2];
};

static __device__ __forceinline__ v8f wmma_bf16(v16bf a, v16bf b, v8f c) {
  // D = A(16x32 bf16) * B(32x16 bf16) + C(16x16 f32)
  return __builtin_amdgcn_wmma_f32_16x16x32_bf16(false, a, false, b, (short)0, c,
                                                 false, false);
}

// Global -> LDS 16B copy. CDNA5 async path (ASYNCcnt) with plain fallback.
static __device__ __forceinline__ void g2l_b128(void* lds_dst, const void* gsrc) {
#if USE_ASYNC_LDS
  unsigned int la = (unsigned int)(unsigned long long)lds_dst;  // low 32b = LDS offset
  asm volatile("global_load_async_to_lds_b128 %0, %1, off"
               :: "v"(la), "v"(gsrc) : "memory");
#else
  *(u32x4*)lds_dst = *(const u32x4*)gsrc;
#endif
}

static __device__ __forceinline__ void wait_async_all() {
#if USE_ASYNC_LDS
#if __has_builtin(__builtin_amdgcn_s_wait_asynccnt)
  __builtin_amdgcn_s_wait_asynccnt(0);
#else
  asm volatile("s_wait_asynccnt 0" ::: "memory");
#endif
#endif
}

// ---------------------------------------------------------------------------
// Kernel 1: generic 16x16-tile WMMA GEMM, one wave per tile. COLS/MODE are
// compile-time so all strided weight loads become immediate-offset b32 loads.
//   MODE 0: keys   = c @ Wk  -> f32  [b*M, K]
//   MODE 1: values = c @ Wv  -> bf16 raw [b*M, V] + bf16 swizzled WMMA-B layout
//   MODE 2: q      = x @ Wq  -> bf16 [b, h, n, k]
// ---------------------------------------------------------------------------
template <int COLS, int MODE>
__global__ __launch_bounds__(32) void proj_kernel(
    const float* __restrict__ A, const float* __restrict__ W,
    float* __restrict__ keysf, __bf16* __restrict__ valsraw,
    __bf16* __restrict__ valssw, __bf16* __restrict__ qws)
{
  const int lane = threadIdx.x & 31;
  const int tr = blockIdx.x, tc = blockIdx.y;
  const int l15 = lane & 15;
  const int kb  = (lane < 16) ? 0 : 8;        // K-half select per A/B layout
  const int rowA = tr * 16 + l15;
  const int col  = tc * 16 + l15;

  const v8f zero8 = {0, 0, 0, 0, 0, 0, 0, 0};
  v8f acc = zero8;

  const float* ap = A + (size_t)rowA * D_ + kb;
  const float* wp = W + (size_t)kb * COLS + col;
  for (int kk = 0; kk < D_; kk += 32) {
    BF16V a, b;
    // A: lane holds row rowA, halves 0..7 = k kb..kb+7, halves 8..15 = +16
    f32x4 a0 = *(const f32x4*)(ap + kk);
    f32x4 a1 = *(const f32x4*)(ap + kk + 4);
    f32x4 a2 = *(const f32x4*)(ap + kk + 16);
    f32x4 a3 = *(const f32x4*)(ap + kk + 20);
#pragma unroll
    for (int j = 0; j < 4; ++j) {
      a.e[j]      = (__bf16)a0[j];
      a.e[4 + j]  = (__bf16)a1[j];
      a.e[8 + j]  = (__bf16)a2[j];
      a.e[12 + j] = (__bf16)a3[j];
    }
    // B: lane holds col, halves 0..7 = k kb..kb+7, halves 8..15 = +16
    const float* wk = wp + (size_t)kk * COLS;
#pragma unroll
    for (int j = 0; j < 8; ++j) {
      b.e[j]     = (__bf16)wk[j * COLS];           // immediate offsets
      b.e[8 + j] = (__bf16)wk[(j + 16) * COLS];
    }
    acc = wmma_bf16(a.v, b.v, acc);
  }

  const int rb = (lane < 16) ? 0 : 8;         // C/D row split at lane 16
#pragma unroll
  for (int r = 0; r < 8; ++r) {
    const int row = tr * 16 + r + rb;
    const int c   = tc * 16 + l15;
    const float val = acc[r];
    if (MODE == 0) {
      keysf[(size_t)row * K_ + c] = val;
    } else if (MODE == 1) {
      valsraw[(size_t)row * V_ + c] = (__bf16)val;
      // Pre-swizzle into per-lane WMMA B-operand layout for the main kernel.
      const int bb  = row >> 11;              // / M_
      const int m   = row & (M_ - 1);
      const int mt  = m >> 5, m32 = m & 31;
      const int vt  = c >> 4, vl = c & 15;
      const int base = m32 & 15;
      const int lp = vl + ((base >= 8) ? 16 : 0);
      const int hp = (base & 7) + ((m32 >= 16) ? 8 : 0);
      valssw[((((size_t)bb * (M_ / 32) + mt) * 8 + vt) * 32 + lp) * 16 + hp] =
          (__bf16)val;
    } else {
      const int bb = row >> 11;               // / N_
      const int n  = row & (N_ - 1);
      const int hh = c >> 4, k = c & 15;
      qws[(((size_t)(bb * H_ + hh) * N_) + n) * K_ + k] = (__bf16)val;
    }
  }
}

// ---------------------------------------------------------------------------
// Kernel 2a: per-(b,k) softmax stats over m (max + sum of exp).
// ---------------------------------------------------------------------------
__global__ __launch_bounds__(256) void softmax_stats_kernel(
    const float* __restrict__ keysf, float* __restrict__ smax,
    float* __restrict__ ssum)
{
  __shared__ float red[256];
  const int bk = blockIdx.x;
  const int b = bk / K_, k = bk - b * K_;
  const int t = threadIdx.x;

  float mx = -3.0e38f;
  for (int m = t; m < M_; m += 256)
    mx = fmaxf(mx, keysf[((size_t)b * M_ + m) * K_ + k]);
  red[t] = mx; __syncthreads();
  for (int s = 128; s > 0; s >>= 1) {
    if (t < s) red[t] = fmaxf(red[t], red[t + s]);
    __syncthreads();
  }
  mx = red[0]; __syncthreads();

  float sm = 0.f;
  for (int m = t; m < M_; m += 256)
    sm += __expf(keysf[((size_t)b * M_ + m) * K_ + k] - mx);
  red[t] = sm; __syncthreads();
  for (int s = 128; s > 0; s >>= 1) {
    if (t < s) red[t] += red[t + s];
    __syncthreads();
  }
  if (t == 0) { smax[bk] = mx; ssum[bk] = red[0]; }
}

// ---------------------------------------------------------------------------
// Kernel 2b: content_lambda partials: CLf[b,k,v] += exp(keys-max)^T @ values.
// Grid (B, M/128); 8 waves, wave -> v-tile. f32 atomics combine m-chunks.
// ---------------------------------------------------------------------------
__global__ __launch_bounds__(256) void content_lambda_kernel(
    const float* __restrict__ keysf, const __bf16* __restrict__ valsraw,
    const float* __restrict__ smax, float* __restrict__ CLf)
{
  const int b = blockIdx.x, mc = blockIdx.y;
  const int w = threadIdx.x >> 5, lane = threadIdx.x & 31;
  const int l15 = lane & 15;
  const int k = l15;
  const int moff = (lane < 16) ? 0 : 8;
  const int v = w * 16 + l15;
  const float mx = smax[b * K_ + k];

  const v8f zero8 = {0, 0, 0, 0, 0, 0, 0, 0};
  v8f acc = zero8;
  for (int it = 0; it < 4; ++it) {
    const int mb = mc * 128 + it * 32;
    const float*  kp = keysf + ((size_t)b * M_ + mb + moff) * K_ + k;
    const __bf16* vp = valsraw + ((size_t)b * M_ + mb + moff) * V_ + v;
    BF16V a, bb;
#pragma unroll
    for (int j = 0; j < 8; ++j) {
      a.e[j]      = (__bf16)__expf(kp[j * K_] - mx);
      a.e[8 + j]  = (__bf16)__expf(kp[(j + 16) * K_] - mx);
      bb.e[j]     = vp[j * V_];
      bb.e[8 + j] = vp[(j + 16) * V_];
    }
    acc = wmma_bf16(a.v, bb.v, acc);
  }
  const int rb = (lane < 16) ? 0 : 8;
#pragma unroll
  for (int r = 0; r < 8; ++r)
    atomicAdd(&CLf[((size_t)b * K_ + r + rb) * V_ + w * 16 + l15], acc[r]);
}

// Kernel 2c: divide by softmax sum, convert to bf16 in [b][v][k] layout.
__global__ __launch_bounds__(256) void cl_finalize_kernel(
    const float* __restrict__ CLf, const float* __restrict__ ssum,
    __bf16* __restrict__ clbf)
{
  const int idx = blockIdx.x * 256 + threadIdx.x;   // over B*V*K
  const int b = idx / (V_ * K_);
  const int rem = idx - b * (V_ * K_);
  const int v = rem / K_, k = rem - v * K_;
  clbf[idx] = (__bf16)(CLf[((size_t)b * K_ + k) * V_ + v] / ssum[b * K_ + k]);
}

// ---------------------------------------------------------------------------
// Kernel 3: fused position+content. Grid (N/16, MSPLIT), 8 waves.
// Per 32-m tile: async-stage E (raw f32) + values (pre-swizzled bf16) into
// double-buffered LDS; stage1 S = q @ E^T (rows = 16 b*h); transpose S via LDS
// to bf16; stage2 acc += S[n x 32m] @ values[32m x 16v] per (b,h,vtile).
// ---------------------------------------------------------------------------
__global__ __launch_bounds__(256) void lambda_main_kernel(
    const float* __restrict__ E, const __bf16* __restrict__ qws,
    const __bf16* __restrict__ valssw, const __bf16* __restrict__ clbf,
    float* __restrict__ out)
{
  __shared__ __align__(16) float  Ebuf[2][16 * 32 * 16];     // 2 x 32 KB
  __shared__ __align__(16) __bf16 Vbuf[2][4 * 8 * 32 * 16];  // 2 x 32 KB
  __shared__ __align__(16) __bf16 Sbuf[16 * 16 * 32];        // 16 KB

  const int tid = threadIdx.x;
  const int lane = tid & 31;
  const int w = tid >> 5;
  const int l15 = lane & 15;
  const int kb = (lane < 16) ? 0 : 8;
  const int n0 = blockIdx.x * 16;
  const int msBase = blockIdx.y * (M_ / MSPLIT);
  const int NT = (M_ / MSPLIT) / 32;
  const int b = w >> 1;                 // this wave's batch (bh pair 2w,2w+1)

  const u32x4 zq = {0, 0, 0, 0};
  const v8f zero8 = {0, 0, 0, 0, 0, 0, 0, 0};

  // Stage-1 A operands (rows = 16 (b,h) combos), one per assigned n; m-invariant.
  const int bq = l15 >> 2, hq = l15 & 3;
  BF16V Aq[2];
#pragma unroll
  for (int p = 0; p < 2; ++p) {
    const int n = n0 + 2 * w + p;
    Aq[p].q[0] = *(const u32x4*)(qws + (((size_t)(bq * H_ + hq) * N_ + n) * K_ + kb));
    Aq[p].q[1] = zq;                    // K 16..31 zero-padded
  }

  // Accumulators: 2 bh x 8 v-tiles; init with content term on m-split 0.
  v8f acc[2][8];
  if (blockIdx.y == 0) {
#pragma unroll
    for (int i = 0; i < 2; ++i) {
      const int hb = (2 * w + i) & 3;
      BF16V Ac;
      Ac.q[0] = *(const u32x4*)(qws + (((size_t)(b * H_ + hb) * N_ + n0 + l15) * K_ + kb));
      Ac.q[1] = zq;
#pragma unroll
      for (int vt = 0; vt < 8; ++vt) {
        BF16V Bc;
        Bc.q[0] = *(const u32x4*)(clbf + ((size_t)b * V_ + vt * 16 + l15) * K_ + kb);
        Bc.q[1] = zq;
        acc[i][vt] = wmma_bf16(Ac.v, Bc.v, zero8);
      }
    }
  } else {
#pragma unroll
    for (int i = 0; i < 2; ++i)
#pragma unroll
      for (int vt = 0; vt < 8; ++vt)
        acc[i][vt] = zero8;
  }

  auto stage = [&](int t, int buf) {
    const int mb = msBase + t * 32;
    const int mt = mb >> 5;
    // E tile: 16n x 32m x 16k f32 (2 KB contiguous per n) -> 8 b128 per thread
#pragma unroll
    for (int j = 0; j < 8; ++j) {
      const int f = tid + j * 256;          // 0..2047 float4 units
      const int nl = f >> 7, r = f & 127;
      const float* src = E + ((size_t)(n0 + nl) * M_ + mb) * K_ + r * 4;
      g2l_b128(&Ebuf[buf][(size_t)nl * 512 + r * 4], src);
    }
    // values tile: pre-swizzled, 8 KB contiguous per b -> 8 b128 per thread
#pragma unroll
    for (int j = 0; j < 8; ++j) {
      const int f = tid + j * 256;
      const int bb2 = f >> 9, r = f & 511;
      const __bf16* src = valssw + ((size_t)(bb2 * (M_ / 32)) + mt) * 4096 + r * 8;
      g2l_b128(&Vbuf[buf][(size_t)bb2 * 4096 + r * 8], src);
    }
  };

  stage(0, 0);
  wait_async_all();
  __syncthreads();

  for (int t = 0; t < NT; ++t) {
    const int buf = t & 1;
    if (t + 1 < NT) stage(t + 1, buf ^ 1);   // overlap with compute

    // Stage 1: S[bh, m] for this wave's 2 n values x 2 m-subtiles.
#pragma unroll
    for (int task = 0; task < 4; ++task) {
      const int p = task >> 1, msub = task & 1;
      const int nl = 2 * w + p;
      const int mloc = msub * 16 + l15;
      const float* e = &Ebuf[buf][((size_t)nl * 32 + mloc) * 16 + kb];
      f32x4 e0 = *(const f32x4*)e;
      f32x4 e1 = *(const f32x4*)(e + 4);
      BF16V Bb;
#pragma unroll
      for (int j = 0; j < 4; ++j) { Bb.e[j] = (__bf16)e0[j]; Bb.e[4 + j] = (__bf16)e1[j]; }
      Bb.q[1] = zq;                         // K 16..31 zero-padded
      v8f d = wmma_bf16(Aq[p].v, Bb.v, zero8);
      const int rb = (lane < 16) ? 0 : 8;
#pragma unroll
      for (int r = 0; r < 8; ++r)           // transpose S into LDS as bf16
        Sbuf[((size_t)(r + rb) * 16 + nl) * 32 + mloc] = (__bf16)d[r];
    }
    __syncthreads();

    // Stage 2: acc[bh][vt] += S[16n x 32m] @ values[32m x 16v]
    BF16V As[2];
#pragma unroll
    for (int i = 0; i < 2; ++i) {
      const __bf16* s = &Sbuf[((size_t)(2 * w + i) * 16 + l15) * 32 + kb];
      As[i].q[0] = *(const u32x4*)s;
      As[i].q[1] = *(const u32x4*)(s + 16);
    }
#pragma unroll
    for (int vt = 0; vt < 8; ++vt) {
      const __bf16* vsrc = &Vbuf[buf][((size_t)b * 8 + vt) * 512 + lane * 16];
      BF16V Bv;
      Bv.q[0] = *(const u32x4*)vsrc;
      Bv.q[1] = *(const u32x4*)(vsrc + 8);
      acc[0][vt] = wmma_bf16(As[0].v, Bv.v, acc[0][vt]);
      acc[1][vt] = wmma_bf16(As[1].v, Bv.v, acc[1][vt]);
    }

    wait_async_all();    // next tile landed (this thread)
    __syncthreads();     // all threads landed; S safe to overwrite
  }

  // Epilogue: combine m-splits via f32 atomics into zeroed output.
  const int rb = (lane < 16) ? 0 : 8;
#pragma unroll
  for (int i = 0; i < 2; ++i) {
    const int hb = (2 * w + i) & 3;
#pragma unroll
    for (int vt = 0; vt < 8; ++vt)
#pragma unroll
      for (int r = 0; r < 8; ++r) {
        const int n = n0 + r + rb;
        const int v = vt * 16 + l15;
        atomicAdd(&out[(((size_t)b * N_ + n) * H_ + hb) * V_ + v], acc[i][vt][r]);
      }
  }
}

// ---------------------------------------------------------------------------
extern "C" void kernel_launch(void* const* d_in, const int* in_sizes, int n_in,
                              void* d_out, int out_size, void* d_ws, size_t ws_size,
                              hipStream_t stream)
{
  (void)in_sizes; (void)n_in; (void)out_size; (void)ws_size;
  const float* x  = (const float*)d_in[0];
  const float* c  = (const float*)d_in[1];
  const float* Wq = (const float*)d_in[2];
  const float* Wk = (const float*)d_in[3];
  const float* Wv = (const float*)d_in[4];
  const float* E  = (const float*)d_in[5];
  float* out = (float*)d_out;

  char* ws = (char*)d_ws;
  size_t off = 0;
  auto alloc = [&](size_t bytes) -> char* {
    char* p = ws + off;
    off = (off + bytes + 255) & ~(size_t)255;
    return p;
  };
  __bf16* qws     = (__bf16*)alloc((size_t)B_ * H_ * N_ * K_ * 2);
  __bf16* valssw  = (__bf16*)alloc((size_t)B_ * M_ * V_ * 2);
  __bf16* valsraw = (__bf16*)alloc((size_t)B_ * M_ * V_ * 2);
  float*  keysf   = (float*) alloc((size_t)B_ * M_ * K_ * 4);
  float*  smax    = (float*) alloc((size_t)B_ * K_ * 4);
  float*  ssum    = (float*) alloc((size_t)B_ * K_ * 4);
  float*  CLf     = (float*) alloc((size_t)B_ * K_ * V_ * 4);
  __bf16* clbf    = (__bf16*)alloc((size_t)B_ * V_ * K_ * 2);

  hipMemsetAsync(CLf, 0, (size_t)B_ * K_ * V_ * 4, stream);
  hipMemsetAsync(out, 0, (size_t)B_ * N_ * D_ * 4, stream);

  proj_kernel<K_, 0><<<dim3((B_ * M_) / 16, K_ / 16), 32, 0, stream>>>(
      c, Wk, keysf, valsraw, valssw, qws);
  proj_kernel<V_, 1><<<dim3((B_ * M_) / 16, V_ / 16), 32, 0, stream>>>(
      c, Wv, keysf, valsraw, valssw, qws);
  proj_kernel<K_ * H_, 2><<<dim3((B_ * N_) / 16, (K_ * H_) / 16), 32, 0, stream>>>(
      x, Wq, keysf, valsraw, valssw, qws);

  softmax_stats_kernel<<<B_ * K_, 256, 0, stream>>>(keysf, smax, ssum);
  content_lambda_kernel<<<dim3(B_, M_ / 128), 256, 0, stream>>>(
      keysf, valsraw, smax, CLf);
  cl_finalize_kernel<<<(B_ * V_ * K_) / 256, 256, 0, stream>>>(CLf, ssum, clbf);

  lambda_main_kernel<<<dim3(N_ / 16, MSPLIT), 256, 0, stream>>>(
      E, qws, valssw, clbf, out);
}
